// TemporalSAE_28724741276350
// MI455X (gfx1250) — matrix operand order repaired and checked by reading, not compiled, not run on hardware
//
#include <hip/hip_runtime.h>

#define D_MODEL    2048
#define N_FEATURES 32768
#define TOPK       64
#define BATCH      8192

// Encoder tiling
#define BM 128
#define BN 128
#define BK 32      // fallback (fp32->bf16 in-loop) K step
#define BKA 64     // async bf16 path K step

typedef __attribute__((ext_vector_type(16))) __bf16 bf16x16;
typedef __attribute__((ext_vector_type(8)))  __bf16 bf16x8;
typedef __attribute__((ext_vector_type(8)))  float  f32x8;
typedef __attribute__((ext_vector_type(4)))  float  f32x4;
typedef __attribute__((ext_vector_type(4)))  unsigned uint4v;

union FragBF16 { bf16x16 v; struct { bf16x8 lo; bf16x8 hi; } p; };
union FU32 { float f; unsigned u; };

__device__ __forceinline__ unsigned short f32_to_bf16_bits(float f) {
  FU32 v; v.f = f;
  unsigned r = v.u + 0x7FFFu + ((v.u >> 16) & 1u);
  return (unsigned short)(r >> 16);
}
__device__ __forceinline__ __bf16 f32_to_bf16_rne(float f) {
  union { unsigned short s; __bf16 b; } o;
  o.s = f32_to_bf16_bits(f);
  return o.b;
}

// Issue one per-lane 16B async global->LDS copy (ASYNCcnt-tracked).
#define ASYNC_B128(ldsoff, goff, sbase)                                   \
  asm volatile("global_load_async_to_lds_b128 %0, %1, %2"                 \
               :: "v"(ldsoff), "v"(goff), "s"(sbase) : "memory")
#define WAIT_ASYNC_8()  asm volatile("s_wait_asynccnt 0x8" ::: "memory")
#define WAIT_ASYNC_0()  asm volatile("s_wait_asynccnt 0x0" ::: "memory")

// ---------------------------------------------------------------------------
// Kernel 0: one-time fp32 -> bf16 conversion (RNE). 8 elems/thread.
// ---------------------------------------------------------------------------
__global__ __launch_bounds__(256)
void to_bf16(const float* __restrict__ in, unsigned short* __restrict__ out, int n)
{
  const int i = (blockIdx.x * 256 + threadIdx.x) * 8;
  if (i >= n) return;
  f32x4 a = *(const f32x4*)(in + i);
  f32x4 b = *(const f32x4*)(in + i + 4);
  uint4v o;
  o.x = (unsigned)f32_to_bf16_bits(a[0]) | ((unsigned)f32_to_bf16_bits(a[1]) << 16);
  o.y = (unsigned)f32_to_bf16_bits(a[2]) | ((unsigned)f32_to_bf16_bits(a[3]) << 16);
  o.z = (unsigned)f32_to_bf16_bits(b[0]) | ((unsigned)f32_to_bf16_bits(b[1]) << 16);
  o.w = (unsigned)f32_to_bf16_bits(b[2]) | ((unsigned)f32_to_bf16_bits(b[3]) << 16);
  *(uint4v*)(out + i) = o;
}

// ---------------------------------------------------------------------------
// Kernel 1a: encoder GEMM from pre-converted bf16, async double-buffered LDS.
//   xb:  [BATCH, D_MODEL] bf16     (WMMA A, native K-contiguous layout)
//   web: [N_FEATURES, D_MODEL] bf16 (WMMA B, native layout)
// Block 128x128, 256 thr = 8 waves (2M x 4N), wave = 64x32 = 4x2 C tiles.
// Per BKA=64 chunk: 8 async b128 copies/thread, 2 k-steps x 8 WMMA/wave.
// ---------------------------------------------------------------------------
__global__ __launch_bounds__(256)
void sae_encoder_async(const unsigned short* __restrict__ xb,
                       const unsigned short* __restrict__ web,
                       const float* __restrict__ be, float* __restrict__ fbuf)
{
  __shared__ __bf16 sA[2][BM * BKA];  // 2 x 16 KB
  __shared__ __bf16 sB[2][BN * BKA];  // 2 x 16 KB

  const int tid   = threadIdx.x;
  const int lane  = tid & 31;
  const int wave  = tid >> 5;
  const int waveM = wave >> 2;    // 0..1
  const int waveN = wave & 3;     // 0..3
  const int bm = blockIdx.x * BM;
  const int bn = blockIdx.y * BN;
  const int mlane = lane & 15;
  const int khalf = lane >> 4;

  f32x8 acc[4][2];
  #pragma unroll
  for (int ms = 0; ms < 4; ++ms)
    #pragma unroll
    for (int ns = 0; ns < 2; ++ns)
      #pragma unroll
      for (int r = 0; r < 8; ++r) acc[ms][ns][r] = 0.0f;

  // Async staging assignment: tile = 128 rows x 128B; 1024 16B units; 4/thread.
  unsigned agoff[4], bgoff[4], ldso[4];
  #pragma unroll
  for (int i = 0; i < 4; ++i) {
    const unsigned u = (unsigned)tid + (unsigned)i * 256u;
    const unsigned r = u >> 3, c = u & 7u;          // row, 16B-col
    agoff[i] = ((bm + r) * (unsigned)D_MODEL + c * 8u) * 2u;  // bytes (sans kc)
    bgoff[i] = ((bn + r) * (unsigned)D_MODEL + c * 8u) * 2u;
    ldso[i]  = r * (BKA * 2u) + c * 16u;                      // bytes in buffer
  }
  const unsigned sAb[2] = { (unsigned)(uintptr_t)&sA[0][0],
                            (unsigned)(uintptr_t)&sA[1][0] };
  const unsigned sBb[2] = { (unsigned)(uintptr_t)&sB[0][0],
                            (unsigned)(uintptr_t)&sB[1][0] };

  // Prefetch chunk 0 into buffer 0
  #pragma unroll
  for (int i = 0; i < 4; ++i) ASYNC_B128(sAb[0] + ldso[i], agoff[i], xb);
  #pragma unroll
  for (int i = 0; i < 4; ++i) ASYNC_B128(sBb[0] + ldso[i], bgoff[i], web);

  const int nchunks = D_MODEL / BKA;  // 32
  for (int ci = 0; ci < nchunks; ++ci) {
    const int cur = ci & 1;
    if (ci + 1 < nchunks) {
      const int nxt = cur ^ 1;
      const unsigned kb = (unsigned)(ci + 1) * (BKA * 2u);
      #pragma unroll
      for (int i = 0; i < 4; ++i) ASYNC_B128(sAb[nxt] + ldso[i], agoff[i] + kb, xb);
      #pragma unroll
      for (int i = 0; i < 4; ++i) ASYNC_B128(sBb[nxt] + ldso[i], bgoff[i] + kb, web);
      WAIT_ASYNC_8();      // drain chunk ci's 8 copies; keep chunk ci+1 in flight
    } else {
      WAIT_ASYNC_0();
    }
    __syncthreads();       // buffer `cur` ready for everyone

    const __bf16* As = &sA[cur][0];
    const __bf16* Bs = &sB[cur][0];
    #pragma unroll
    for (int s = 0; s < 2; ++s) {   // two 16x16x32 k-steps inside BKA=64
      FragBF16 af[4], bf[2];
      #pragma unroll
      for (int ms = 0; ms < 4; ++ms) {
        const __bf16* base = As + (waveM * 64 + ms * 16 + mlane) * BKA + s * 32 + khalf * 8;
        af[ms].p.lo = *(const bf16x8*)(base);
        af[ms].p.hi = *(const bf16x8*)(base + 16);
      }
      #pragma unroll
      for (int ns = 0; ns < 2; ++ns) {
        const __bf16* base = Bs + (waveN * 32 + ns * 16 + mlane) * BKA + s * 32 + khalf * 8;
        bf[ns].p.lo = *(const bf16x8*)(base);
        bf[ns].p.hi = *(const bf16x8*)(base + 16);
      }
      #pragma unroll
      for (int ms = 0; ms < 4; ++ms)
        #pragma unroll
        for (int ns = 0; ns < 2; ++ns)
          acc[ms][ns] = __builtin_amdgcn_wmma_f32_16x16x32_bf16(
              false, af[ms].v, false, bf[ns].v,
              (short)0, acc[ms][ns], false, false);
    }
    __syncthreads();       // done reading `cur`; it may be refilled at ci+2
  }

  const int mhi = khalf * 8;
  #pragma unroll
  for (int ns = 0; ns < 2; ++ns) {
    const int col = bn + waveN * 32 + ns * 16 + mlane;
    const float bias = be[col];
    #pragma unroll
    for (int ms = 0; ms < 4; ++ms) {
      #pragma unroll
      for (int r = 0; r < 8; ++r) {
        const int rowo = bm + waveM * 64 + ms * 16 + mhi + r;
        float v = acc[ms][ns][r] + bias;
        v = v > 0.0f ? v : 0.0f;
        fbuf[(size_t)rowo * N_FEATURES + col] = v;
      }
    }
  }
}

// ---------------------------------------------------------------------------
// Kernel 1b: fallback encoder (fp32 inputs, convert at staging). Round-1 code.
// ---------------------------------------------------------------------------
__global__ __launch_bounds__(256)
void sae_encoder(const float* __restrict__ x, const float* __restrict__ We,
                 const float* __restrict__ be, float* __restrict__ fbuf)
{
  __shared__ __bf16 sA[BM * BK];
  __shared__ __bf16 sB[BN * BK];

  const int tid   = threadIdx.x;
  const int lane  = tid & 31;
  const int wave  = tid >> 5;
  const int waveM = wave >> 2;
  const int waveN = wave & 3;
  const int bm = blockIdx.x * BM;
  const int bn = blockIdx.y * BN;

  f32x8 acc[4][2];
  #pragma unroll
  for (int ms = 0; ms < 4; ++ms)
    #pragma unroll
    for (int ns = 0; ns < 2; ++ns)
      #pragma unroll
      for (int r = 0; r < 8; ++r) acc[ms][ns][r] = 0.0f;

  const int srow = tid >> 1;
  const int kseg = (tid & 1) * 16;
  const int mlane = lane & 15;
  const int khalf = lane >> 4;

  for (int kc = 0; kc < D_MODEL; kc += BK) {
    const float* ga = x  + (size_t)(bm + srow) * D_MODEL + kc + kseg;
    const float* gb = We + (size_t)(bn + srow) * D_MODEL + kc + kseg;
    f32x4 va[4], vb[4];
    #pragma unroll
    for (int i = 0; i < 4; ++i) {
      va[i] = *(const f32x4*)(ga + 4 * i);
      vb[i] = *(const f32x4*)(gb + 4 * i);
    }

    __syncthreads();
    #pragma unroll
    for (int i = 0; i < 4; ++i)
      #pragma unroll
      for (int j = 0; j < 4; ++j) {
        sA[srow * BK + kseg + 4 * i + j] = f32_to_bf16_rne(va[i][j]);
        sB[srow * BK + kseg + 4 * i + j] = f32_to_bf16_rne(vb[i][j]);
      }
    __syncthreads();

    FragBF16 af[4], bf[2];
    #pragma unroll
    for (int ms = 0; ms < 4; ++ms) {
      const __bf16* base = &sA[(waveM * 64 + ms * 16 + mlane) * BK + khalf * 8];
      af[ms].p.lo = *(const bf16x8*)(base);
      af[ms].p.hi = *(const bf16x8*)(base + 16);
    }
    #pragma unroll
    for (int ns = 0; ns < 2; ++ns) {
      const __bf16* base = &sB[(waveN * 32 + ns * 16 + mlane) * BK + khalf * 8];
      bf[ns].p.lo = *(const bf16x8*)(base);
      bf[ns].p.hi = *(const bf16x8*)(base + 16);
    }
    #pragma unroll
    for (int ms = 0; ms < 4; ++ms)
      #pragma unroll
      for (int ns = 0; ns < 2; ++ns)
        acc[ms][ns] = __builtin_amdgcn_wmma_f32_16x16x32_bf16(
            false, af[ms].v, false, bf[ns].v,
            (short)0, acc[ms][ns], false, false);
  }

  const int mhi = khalf * 8;
  #pragma unroll
  for (int ns = 0; ns < 2; ++ns) {
    const int col = bn + waveN * 32 + ns * 16 + mlane;
    const float bias = be[col];
    #pragma unroll
    for (int ms = 0; ms < 4; ++ms)
      #pragma unroll
      for (int r = 0; r < 8; ++r) {
        const int rowo = bm + waveM * 64 + ms * 16 + mhi + r;
        float v = acc[ms][ns][r] + bias;
        v = v > 0.0f ? v : 0.0f;
        fbuf[(size_t)rowo * N_FEATURES + col] = v;
      }
  }
}

// ---------------------------------------------------------------------------
// Kernel 2: per-row top-64 via 3-level radix select on float bits.
// ---------------------------------------------------------------------------
__global__ __launch_bounds__(256)
void sae_topk(float* __restrict__ fbuf, int* __restrict__ nz_idx,
              float* __restrict__ nz_val)
{
  __shared__ unsigned hist[2048];
  __shared__ unsigned sh_prefix, sh_above, sh_eq, sh_out;

  const int row = blockIdx.x;
  const int tid = threadIdx.x;
  float* frow = fbuf + (size_t)row * N_FEATURES;

  unsigned prefix = 0, above = 0;

  #pragma unroll 1
  for (int level = 0; level < 3; ++level) {
    const int shift = (level == 0) ? 21 : (level == 1) ? 10 : 0;
    const int bins  = (level == 2) ? 1024 : 2048;

    for (int i = tid; i < bins; i += 256) hist[i] = 0u;
    __syncthreads();

    for (int i = tid; i < N_FEATURES; i += 256) {
      FU32 v; v.f = frow[i];
      bool in_bucket = (level == 0) ||
                       ((v.u >> (shift + 11)) == (prefix >> (shift + 11)));
      if (in_bucket) atomicAdd(&hist[(v.u >> shift) & (unsigned)(bins - 1)], 1u);
    }
    __syncthreads();

    if (tid == 0) {
      unsigned acc = above;
      int b = bins - 1;
      for (; b > 0; --b) {
        if (acc + hist[b] >= TOPK) break;
        acc += hist[b];
      }
      sh_above  = acc;
      sh_prefix = prefix | ((unsigned)b << shift);
    }
    __syncthreads();
    prefix = sh_prefix;
    above  = sh_above;
    __syncthreads();
  }

  if (tid == 0) { sh_eq = 0u; sh_out = 0u; }
  __syncthreads();

  const unsigned T = prefix, n_gt = above;
  for (int i = tid; i < N_FEATURES; i += 256) {
    FU32 v; v.f = frow[i];
    bool keep;
    if (v.u > T)       keep = true;
    else if (v.u == T) keep = (n_gt + atomicAdd(&sh_eq, 1u)) < TOPK;
    else               keep = false;
    if (keep) {
      unsigned slot = atomicAdd(&sh_out, 1u);
      if (slot < TOPK) {
        nz_idx[(size_t)row * TOPK + slot] = i;
        nz_val[(size_t)row * TOPK + slot] = v.f;
      }
    } else {
      frow[i] = 0.0f;
    }
  }
}

// ---------------------------------------------------------------------------
// Kernel 3: sparse decoder  x_hat[row,:] = sum_j val_j * W_dec[:, idx_j] + b_dec
// ---------------------------------------------------------------------------
__global__ __launch_bounds__(256)
void sae_decoder(const int* __restrict__ nz_idx, const float* __restrict__ nz_val,
                 const float* __restrict__ Wd, const float* __restrict__ bd,
                 float* __restrict__ xhat)
{
  __shared__ int   sidx[TOPK];
  __shared__ float sval[TOPK];

  const int row = blockIdx.x;
  const int tid = threadIdx.x;

  if (tid < TOPK) {
    sidx[tid] = nz_idx[(size_t)row * TOPK + tid];
    sval[tid] = nz_val[(size_t)row * TOPK + tid];
  }
  __syncthreads();

  #pragma unroll
  for (int dd = 0; dd < D_MODEL / 256; ++dd) {
    const int d = dd * 256 + tid;
    float acc = bd[d];
    const float* wrow = Wd + (size_t)d * N_FEATURES;
    #pragma unroll 8
    for (int j = 0; j < TOPK; ++j) acc += sval[j] * wrow[sidx[j]];
    xhat[(size_t)row * D_MODEL + d] = acc;
  }
}

// ---------------------------------------------------------------------------
extern "C" void kernel_launch(void* const* d_in, const int* in_sizes, int n_in,
                              void* d_out, int out_size, void* d_ws, size_t ws_size,
                              hipStream_t stream)
{
  const float* x  = (const float*)d_in[0];
  const float* We = (const float*)d_in[1];
  const float* be = (const float*)d_in[2];
  const float* Wd = (const float*)d_in[3];
  const float* bd = (const float*)d_in[4];

  float* xhat = (float*)d_out;
  float* fbuf = (float*)d_out + (size_t)BATCH * D_MODEL;

  const size_t nz_i_bytes = (size_t)BATCH * TOPK * sizeof(int);
  const size_t nz_v_bytes = (size_t)BATCH * TOPK * sizeof(float);
  const size_t xb_bytes   = (size_t)BATCH * D_MODEL * 2;
  const size_t web_bytes  = (size_t)N_FEATURES * D_MODEL * 2;

  int*   nz_idx = (int*)d_ws;
  float* nz_val = (float*)((char*)d_ws + nz_i_bytes);

  dim3 egrid(BATCH / BM, N_FEATURES / BN);
  if (ws_size >= nz_i_bytes + nz_v_bytes + xb_bytes + web_bytes) {
    // bf16 pre-conversion + async-copy encoder (preferred path)
    unsigned short* xbf  = (unsigned short*)((char*)d_ws + nz_i_bytes + nz_v_bytes);
    unsigned short* webf = (unsigned short*)((char*)xbf + xb_bytes);
    const int nx = BATCH * D_MODEL;
    const int nw = N_FEATURES * D_MODEL;
    to_bf16<<<nx / (256 * 8), 256, 0, stream>>>(x, xbf, nx);
    to_bf16<<<nw / (256 * 8), 256, 0, stream>>>(We, webf, nw);
    sae_encoder_async<<<egrid, 256, 0, stream>>>(xbf, webf, be, fbuf);
  } else {
    sae_encoder<<<egrid, 256, 0, stream>>>(x, We, be, fbuf);
  }
  sae_topk<<<BATCH, 256, 0, stream>>>(fbuf, nz_idx, nz_val);
  sae_decoder<<<BATCH, 256, 0, stream>>>(nz_idx, nz_val, Wd, bd, xhat);
}
